// MultiHeadAttention_54047868453298
// MI455X (gfx1250) — matrix operand order repaired and checked by reading, not compiled, not run on hardware
//
#include <hip/hip_runtime.h>

// ---------------------------------------------------------------------------
// MHA forward for MI455X (gfx1250), wave32, bf16 WMMA (v_wmma_f32_16x16x32_bf16)
// B=4, S=2048, D=1024, H=16, HD=64
// ---------------------------------------------------------------------------

typedef __bf16 bf16_t;
typedef __attribute__((ext_vector_type(16))) __bf16 bf16x16;
typedef __attribute__((ext_vector_type(8)))  __bf16 bf16x8;
typedef __attribute__((ext_vector_type(8)))  float  f32x8;
typedef __attribute__((ext_vector_type(4)))  float  f32x4;

__device__ __forceinline__ bf16x16 frag2x8(const bf16_t* p0, const bf16_t* p1) {
    bf16x8 lo = *(const bf16x8*)p0;
    bf16x8 hi = *(const bf16x8*)p1;
    return __builtin_shufflevector(lo, hi, 0,1,2,3,4,5,6,7,8,9,10,11,12,13,14,15);
}

__device__ __forceinline__ f32x8 wmma_bf16(bf16x16 a, bf16x16 b, f32x8 c) {
    return __builtin_amdgcn_wmma_f32_16x16x32_bf16(false, a, false, b, (short)0, c,
                                                   false, false);
}

// Async global -> LDS copy of 16 bytes (CDNA5 GLOBAL_LOAD_ASYNC_TO_LDS_B128).
// dst: shared-memory pointer (low 32 bits of the flat address are the
// wave-relative LDS byte address per the aperture mapping); src: global ptr.
__device__ __forceinline__ void async_cp16(bf16_t* dst, const bf16_t* src) {
    asm volatile("global_load_async_to_lds_b128 %0, %1, off"
                 :: "v"((unsigned)(unsigned long long)dst), "v"(src) : "memory");
}

// ---------------------------------------------------------------------------
// Kernel 0: bulk fp32 -> bf16 conversion (x), 8 elements / thread.
// ---------------------------------------------------------------------------
__global__ __launch_bounds__(256) void mha_cvt_kernel(const float* __restrict__ X,
                                                      bf16_t* __restrict__ Xb) {
    const size_t i = ((size_t)blockIdx.x * 256 + threadIdx.x) * 8;
    f32x4 a = *(const f32x4*)(X + i);
    f32x4 b = *(const f32x4*)(X + i + 4);
    bf16x8 o;
    o[0] = (bf16_t)a.x; o[1] = (bf16_t)a.y; o[2] = (bf16_t)a.z; o[3] = (bf16_t)a.w;
    o[4] = (bf16_t)b.x; o[5] = (bf16_t)b.y; o[6] = (bf16_t)b.z; o[7] = (bf16_t)b.w;
    *(bf16x8*)(Xb + i) = o;
}

// ---------------------------------------------------------------------------
// Kernel 1: weight convert + transpose.  W fp32 row-major [k][n] -> Wt bf16 [n][k]
// ---------------------------------------------------------------------------
__global__ __launch_bounds__(256) void mha_wtrans_kernel(const float* __restrict__ W,
                                                         bf16_t* __restrict__ Wt) {
    __shared__ float tile[32][33];
    const int bx = blockIdx.x * 32;   // n base
    const int by = blockIdx.y * 32;   // k base
    const int tx = threadIdx.x;       // 0..31
    const int ty = threadIdx.y;       // 0..7
    #pragma unroll
    for (int i = 0; i < 32; i += 8)
        tile[ty + i][tx] = W[(size_t)(by + ty + i) * 1024 + bx + tx];
    __syncthreads();
    #pragma unroll
    for (int i = 0; i < 32; i += 8)
        Wt[(size_t)(bx + ty + i) * 1024 + by + tx] = (bf16_t)tile[tx][ty + i];
}

// ---------------------------------------------------------------------------
// Kernel 2: GEMM  C[8192,1024] = A(bf16)[8192,1024] @ Wt^T (+bias), bf16 WMMA.
//   mode 0: out bf16 head layout   [b,h,s,hd]   (Q, K)
//   mode 1: out bf16 head-T layout [b,h,hd,s]   (V)
//   mode 2: out fp32 plain [M,1024]             (final projection)
// Block: 256 threads = 8 waves (2x4), tile 128x128, BK=32.
// Double-buffered LDS filled by async direct-to-LDS loads (ASYNCcnt).
// ---------------------------------------------------------------------------
__global__ __launch_bounds__(256) void mha_gemm_kernel(const bf16_t* __restrict__ A,
                                                       const bf16_t* __restrict__ Bt,
                                                       const float* __restrict__ bias,
                                                       void* __restrict__ Cout,
                                                       int mode) {
    __shared__ bf16_t As[2][128 * 40];  // padded stride 40 (80B rows, 16B aligned)
    __shared__ bf16_t Bs[2][128 * 40];  // 40 KB total

    const int tid  = threadIdx.x;
    const int lane = tid & 31;
    const int wave = tid >> 5;
    const int hf   = lane >> 4;       // which 16-lane half
    const int l16  = lane & 15;
    const int wm   = wave >> 2;       // 0..1 -> 64 rows
    const int wn   = wave & 3;        // 0..3 -> 32 cols
    const int row0 = blockIdx.y * 128;
    const int col0 = blockIdx.x * 128;

    // staging coordinates: two 16B chunks per thread per tile
    const int e0 = tid * 8,         r0 = e0 >> 5, c0 = e0 & 31;
    const int e1 = (tid + 256) * 8, r1 = e1 >> 5, c1 = e1 & 31;
    const bf16_t* Aa0 = A  + (size_t)(row0 + r0) * 1024 + c0;
    const bf16_t* Aa1 = A  + (size_t)(row0 + r1) * 1024 + c1;
    const bf16_t* Ba0 = Bt + (size_t)(col0 + r0) * 1024 + c0;
    const bf16_t* Ba1 = Bt + (size_t)(col0 + r1) * 1024 + c1;

    f32x8 acc[4][2];
    #pragma unroll
    for (int mi = 0; mi < 4; ++mi)
        #pragma unroll
        for (int ni = 0; ni < 2; ++ni)
            #pragma unroll
            for (int v = 0; v < 8; ++v) acc[mi][ni][v] = 0.0f;

    // prologue: async-fill buffer 0 with the k0 = 0 tile
    async_cp16(&As[0][r0 * 40 + c0], Aa0);
    async_cp16(&As[0][r1 * 40 + c1], Aa1);
    async_cp16(&Bs[0][r0 * 40 + c0], Ba0);
    async_cp16(&Bs[0][r1 * 40 + c1], Ba1);

    for (int k0 = 0; k0 < 1024; k0 += 32) {
        const int buf = (k0 >> 5) & 1;
        // ---- issue async fill of the other buffer (overlaps with WMMAs) ----
        if (k0 + 32 < 1024) {
            const int kn = k0 + 32, nb = buf ^ 1;
            async_cp16(&As[nb][r0 * 40 + c0], Aa0 + kn);
            async_cp16(&As[nb][r1 * 40 + c1], Aa1 + kn);
            async_cp16(&Bs[nb][r0 * 40 + c0], Ba0 + kn);
            async_cp16(&Bs[nb][r1 * 40 + c1], Ba1 + kn);
            asm volatile("s_wait_asynccnt 0x4" ::: "memory");  // current tile landed
        } else {
            asm volatile("s_wait_asynccnt 0x0" ::: "memory");
        }
        __syncthreads();

        // ---- fragments + 8 WMMAs ----
        bf16x16 af[4], bfr[2];
        #pragma unroll
        for (int mi = 0; mi < 4; ++mi) {
            const bf16_t* base = &As[buf][(wm * 64 + mi * 16 + l16) * 40];
            af[mi] = frag2x8(base + hf * 8, base + 16 + hf * 8);
        }
        #pragma unroll
        for (int ni = 0; ni < 2; ++ni) {
            const bf16_t* base = &Bs[buf][(wn * 32 + ni * 16 + l16) * 40 + hf * 8];
            bfr[ni] = frag2x8(base, base + 16);
        }
        #pragma unroll
        for (int mi = 0; mi < 4; ++mi)
            #pragma unroll
            for (int ni = 0; ni < 2; ++ni)
                acc[mi][ni] = wmma_bf16(af[mi], bfr[ni], acc[mi][ni]);
        __syncthreads();  // protect buffer reuse on the next iteration
    }

    // ---- epilogue ----
    #pragma unroll
    for (int mi = 0; mi < 4; ++mi)
        #pragma unroll
        for (int ni = 0; ni < 2; ++ni) {
            const int gcol = col0 + wn * 32 + ni * 16 + l16;
            const float bv = bias[gcol];
            #pragma unroll
            for (int v = 0; v < 8; ++v) {
                const int grow = row0 + wm * 64 + mi * 16 + hf * 8 + v;
                const float val = acc[mi][ni][v] + bv;
                if (mode == 2) {
                    ((float*)Cout)[(size_t)grow * 1024 + gcol] = val;
                } else {
                    const int b = grow >> 11, s = grow & 2047;
                    const int h = gcol >> 6,  hd = gcol & 63;
                    const size_t bh = (size_t)(b * 16 + h);
                    if (mode == 0)
                        ((bf16_t*)Cout)[(bh * 2048 + s) * 64 + hd] = (bf16_t)val;
                    else
                        ((bf16_t*)Cout)[(bh * 64 + hd) * 2048 + s] = (bf16_t)val;
                }
            }
        }
}

// ---------------------------------------------------------------------------
// Kernel 3: flash attention.  grid (S/64, B*H), block 128 = 4 waves.
// Each wave owns 16 query rows, streams keys in chunks of 64 with online
// softmax; P re-laid-out C->A through per-wave LDS; out -> AO bf16 [B,S,D].
// ---------------------------------------------------------------------------
__global__ __launch_bounds__(128) void mha_attn_kernel(const bf16_t* __restrict__ Qb,
                                                       const bf16_t* __restrict__ Kb,
                                                       const bf16_t* __restrict__ Vt,
                                                       const unsigned char* __restrict__ mask,
                                                       bf16_t* __restrict__ AO) {
    __shared__ bf16_t P[4][16 * 64];  // per-wave P staging (2 KB each)

    const int tid  = threadIdx.x;
    const int lane = tid & 31;
    const int wave = tid >> 5;
    const int hf   = lane >> 4;
    const int l16  = lane & 15;
    const int bh   = blockIdx.y;           // b*16 + h
    const int b    = bh >> 4;
    const int h    = bh & 15;
    const int q0   = blockIdx.x * 64 + wave * 16;
    const float scale = 0.125f;            // HD^-0.5

    // Q fragments: A-layout, 16 rows x 64 hd -> two K=32 fragments
    const bf16_t* qrow = Qb + ((size_t)bh * 2048 + q0 + l16) * 64;
    bf16x16 aq[2];
    #pragma unroll
    for (int j = 0; j < 2; ++j)
        aq[j] = frag2x8(qrow + 32 * j + hf * 8, qrow + 32 * j + 16 + hf * 8);

    f32x8 o[4];
    #pragma unroll
    for (int t = 0; t < 4; ++t)
        #pragma unroll
        for (int v = 0; v < 8; ++v) o[t][v] = 0.0f;
    float mrun[8], lrun[8];
    #pragma unroll
    for (int v = 0; v < 8; ++v) { mrun[v] = -1.0e30f; lrun[v] = 0.0f; }

    for (int kc = 0; kc < 2048; kc += 64) {
        // ---- prefetch next chunk of K / V into cache (global_prefetch_b8) ----
        if (kc + 64 < 2048) {
            __builtin_prefetch(Kb + ((size_t)bh * 2048 + kc + 64 + lane * 2) * 64, 0, 1);
            __builtin_prefetch(Vt + ((size_t)bh * 64 + lane * 2) * 2048 + kc + 64, 0, 1);
        }

        // ---- scores: 16 q x 64 keys (four 16x16 tiles, K-dim = HD = 64) ----
        f32x8 s[4];
        #pragma unroll
        for (int t = 0; t < 4; ++t) {
            #pragma unroll
            for (int v = 0; v < 8; ++v) s[t][v] = 0.0f;
            const bf16_t* krow = Kb + ((size_t)bh * 2048 + kc + t * 16 + l16) * 64;
            #pragma unroll
            for (int j = 0; j < 2; ++j) {
                bf16x16 kf = frag2x8(krow + 32 * j + hf * 8,
                                     krow + 32 * j + 16 + hf * 8);
                s[t] = wmma_bf16(aq[j], kf, s[t]);
            }
            const float madd = mask[b * 2048 + kc + t * 16 + l16] ? -1.0e30f : 0.0f;
            #pragma unroll
            for (int v = 0; v < 8; ++v) s[t][v] = s[t][v] * scale + madd;
        }

        // ---- online softmax statistics (row = v + 8*hf, cols across 16 lanes) ----
        float tmax[8];
        #pragma unroll
        for (int v = 0; v < 8; ++v)
            tmax[v] = fmaxf(fmaxf(s[0][v], s[1][v]), fmaxf(s[2][v], s[3][v]));
        #pragma unroll
        for (int off = 1; off <= 8; off <<= 1)
            #pragma unroll
            for (int v = 0; v < 8; ++v)
                tmax[v] = fmaxf(tmax[v], __shfl_xor(tmax[v], off, 32));

        float al[8];
        #pragma unroll
        for (int v = 0; v < 8; ++v) {
            const float mn = fmaxf(mrun[v], tmax[v]);
            al[v] = __expf(mrun[v] - mn);
            mrun[v] = mn;
        }
        float p[4][8], rs[8];
        #pragma unroll
        for (int v = 0; v < 8; ++v) {
            p[0][v] = __expf(s[0][v] - mrun[v]);
            p[1][v] = __expf(s[1][v] - mrun[v]);
            p[2][v] = __expf(s[2][v] - mrun[v]);
            p[3][v] = __expf(s[3][v] - mrun[v]);
            rs[v] = (p[0][v] + p[1][v]) + (p[2][v] + p[3][v]);
        }
        #pragma unroll
        for (int off = 1; off <= 8; off <<= 1)
            #pragma unroll
            for (int v = 0; v < 8; ++v) rs[v] += __shfl_xor(rs[v], off, 32);
        #pragma unroll
        for (int v = 0; v < 8; ++v) lrun[v] = lrun[v] * al[v] + rs[v];
        #pragma unroll
        for (int t = 0; t < 4; ++t)
            #pragma unroll
            for (int v = 0; v < 8; ++v) o[t][v] *= al[v];

        // ---- C-layout -> A-layout for P via LDS (same wave, ds wait only) ----
        bf16_t* pw = &P[wave][0];
        #pragma unroll
        for (int t = 0; t < 4; ++t)
            #pragma unroll
            for (int v = 0; v < 8; ++v)
                pw[(v + hf * 8) * 64 + t * 16 + l16] = (bf16_t)p[t][v];
        asm volatile("s_wait_dscnt 0" ::: "memory");
        bf16x16 pa[2];
        #pragma unroll
        for (int j = 0; j < 2; ++j)
            pa[j] = frag2x8(pw + l16 * 64 + 32 * j + hf * 8,
                            pw + l16 * 64 + 32 * j + 16 + hf * 8);

        // ---- O += P @ V  (V stored [bh][hd][s]: K-contiguous B fragments) ----
        #pragma unroll
        for (int ht = 0; ht < 4; ++ht) {
            #pragma unroll
            for (int j = 0; j < 2; ++j) {
                const bf16_t* vrow = Vt + ((size_t)bh * 64 + ht * 16 + l16) * 2048 +
                                     kc + 32 * j + hf * 8;
                bf16x16 vf = frag2x8(vrow, vrow + 16);
                o[ht] = wmma_bf16(pa[j], vf, o[ht]);
            }
        }
    }

    // ---- normalize and write AO bf16 [B,S,D] (heads merged) ----
    #pragma unroll
    for (int v = 0; v < 8; ++v) {
        const float inv = 1.0f / lrun[v];
        const size_t rowoff = ((size_t)b * 2048 + q0 + hf * 8 + v) * 1024 + h * 64;
        #pragma unroll
        for (int ht = 0; ht < 4; ++ht)
            AO[rowoff + ht * 16 + l16] = (bf16_t)(o[ht][v] * inv);
    }
}

// ---------------------------------------------------------------------------
// Launch
// ---------------------------------------------------------------------------
extern "C" void kernel_launch(void* const* d_in, const int* in_sizes, int n_in,
                              void* d_out, int out_size, void* d_ws, size_t ws_size,
                              hipStream_t stream) {
    (void)in_sizes; (void)n_in; (void)out_size; (void)ws_size;

    const float* x             = (const float*)d_in[0];
    const unsigned char* mask  = (const unsigned char*)d_in[1];
    const float* Wq = (const float*)d_in[2]; const float* bq = (const float*)d_in[3];
    const float* Wk = (const float*)d_in[4]; const float* bk = (const float*)d_in[5];
    const float* Wv = (const float*)d_in[6]; const float* bv = (const float*)d_in[7];
    const float* Wo = (const float*)d_in[8]; const float* bo = (const float*)d_in[9];

    char* ws = (char*)d_ws;
    const size_t MB = 1ull << 20;
    bf16_t* Wtq = (bf16_t*)(ws + 0 * MB);   // 2 MB each
    bf16_t* Wtk = (bf16_t*)(ws + 2 * MB);
    bf16_t* Wtv = (bf16_t*)(ws + 4 * MB);
    bf16_t* Wto = (bf16_t*)(ws + 6 * MB);
    bf16_t* xb  = (bf16_t*)(ws + 8 * MB);   // 16 MB each
    bf16_t* Qb  = (bf16_t*)(ws + 24 * MB);
    bf16_t* Kb  = (bf16_t*)(ws + 40 * MB);
    bf16_t* Vt  = (bf16_t*)(ws + 56 * MB);
    bf16_t* AO  = (bf16_t*)(ws + 72 * MB);  // ends at 88 MB

    // x -> bf16 (8M elements, 8 per thread)
    mha_cvt_kernel<<<4096, 256, 0, stream>>>(x, xb);

    dim3 tgrid(32, 32), tblk(32, 8);
    mha_wtrans_kernel<<<tgrid, tblk, 0, stream>>>(Wq, Wtq);
    mha_wtrans_kernel<<<tgrid, tblk, 0, stream>>>(Wk, Wtk);
    mha_wtrans_kernel<<<tgrid, tblk, 0, stream>>>(Wv, Wtv);
    mha_wtrans_kernel<<<tgrid, tblk, 0, stream>>>(Wo, Wto);

    dim3 ggrid(8, 64);  // N/128, M/128
    mha_gemm_kernel<<<ggrid, 256, 0, stream>>>(xb, Wtq, bq, Qb, 0);
    mha_gemm_kernel<<<ggrid, 256, 0, stream>>>(xb, Wtk, bk, Kb, 0);
    mha_gemm_kernel<<<ggrid, 256, 0, stream>>>(xb, Wtv, bv, Vt, 1);

    mha_attn_kernel<<<dim3(32, 64), 128, 0, stream>>>(Qb, Kb, Vt, mask, AO);

    mha_gemm_kernel<<<ggrid, 256, 0, stream>>>(AO, Wto, bo, d_out, 2);
}
